// LSTM_26285199851738
// MI455X (gfx1250) — matrix operand order repaired and checked by reading, not compiled, not run on hardware
//
#include <hip/hip_runtime.h>
#include <hip/hip_bf16.h>
#include <math.h>

// ---------------------------------------------------------------------------
// CDNA5 (gfx1250) LSTM: bf16 WMMA (wave32), async-to-LDS gate prefetch,
// 16-WGP persistent recurrence with column-ownership so the cell update is
// workgroup-local and only h (16KB, L2-hot) crosses workgroups per step.
//   B=32, T=2048, D=256, H=256, A=18
// ---------------------------------------------------------------------------

typedef __bf16 bf16_t;
typedef __attribute__((ext_vector_type(16))) __bf16 v16bf;
typedef __attribute__((ext_vector_type(8)))  float  v8f;

#define B_   32
#define T_   2048
#define D_   256
#define H_   256
#define G4   1024   // 4*H
#define A_   18
#define NWG  16     // recurrence workgroups (persistent, grid-barriered)

static __device__ __forceinline__ float sigmoidf_(float x) {
    return 1.0f / (1.0f + __expf(-x));
}

// --- A-fragment (16x32, bf16) from an f32 row-major matrix -----------------
// ISA 7.12.2 (16-bit A 16x32): lanes 0-15 -> M=lane, K={0..7,16..23};
// lanes 16-31 -> M=lane-16, K={8..15,24..31}.
static __device__ __forceinline__ v16bf load_a_f32(const float* base, int ld,
                                                   int m0, int kbase, int lane) {
    const int row   = m0 + (lane & 15);
    const int khalf = (lane >> 4) & 1;
    const float* p  = base + (size_t)row * ld + kbase + khalf * 8;
    v16bf a;
#pragma unroll
    for (int i = 0; i < 8; ++i) {
        a[i]     = (__bf16)p[i];
        a[i + 8] = (__bf16)p[16 + i];
    }
    return a;
}

// Same A layout, bf16 source (L2-hot h buffer).
static __device__ __forceinline__ v16bf load_a_bf16(const bf16_t* base, int ld,
                                                    int m0, int kbase, int lane) {
    const int row   = m0 + (lane & 15);
    const int khalf = (lane >> 4) & 1;
    const bf16_t* p = base + (size_t)row * ld + kbase + khalf * 8;
    v16bf a;
#pragma unroll
    for (int i = 0; i < 8; ++i) {
        a[i]     = p[i];
        a[i + 8] = p[16 + i];
    }
    return a;
}

// --- B-fragment (32x16, bf16) from fragment-native layout ------------------
// Weights pre-packed as frag[n][kc][32] so each lane is one contiguous 32B
// load: lanes 0-15 -> N=lane, K={0..7,16..23}; lanes 16-31 -> N=lane-16, K={8..15,24..31}.
static __device__ __forceinline__ v16bf load_b_frag(const bf16_t* frag, int n0,
                                                    int kc, int lane) {
    const int n     = n0 + (lane & 15);
    const int khalf = (lane >> 4) & 1;
    const bf16_t* p = frag + (((size_t)n * 8 + kc) * 32) + khalf * 16;
    return *(const v16bf*)p;
}

static __device__ __forceinline__ v8f wmma_bf16(v16bf a, v16bf b, v8f c) {
    return __builtin_amdgcn_wmma_f32_16x16x32_bf16(
        false, a, false, b, (short)0, c, false, false);
}

// --- CDNA5 async global->LDS (ASYNCcnt path), inline asm per bridge doc ----
static __device__ __forceinline__ void async_g2l_b128(void* lds_ptr, const void* gptr) {
    const unsigned int loff = (unsigned int)(uintptr_t)lds_ptr;  // addr[31:0] = LDS offset
    asm volatile("global_load_async_to_lds_b128 %0, %1, off"
                 :: "v"(loff), "v"(gptr) : "memory");
}
static __device__ __forceinline__ void wait_async0() {
    asm volatile("s_wait_asynccnt 0x0" ::: "memory");
}

// --- persistent-grid barrier (monotonic counter, agent scope) --------------
static __device__ __forceinline__ void grid_barrier(int* cnt, int target) {
    __syncthreads();
    if (threadIdx.x == 0) {
        __hip_atomic_fetch_add(cnt, 1, __ATOMIC_ACQ_REL, __HIP_MEMORY_SCOPE_AGENT);
        while (__hip_atomic_load(cnt, __ATOMIC_ACQUIRE, __HIP_MEMORY_SCOPE_AGENT) < target)
            __builtin_amdgcn_s_sleep(2);
    }
    __syncthreads();
}

// ===========================================================================
// Kernel 1: repack weights -> bf16 fragment layout; zero the grid-barrier ctr
// ===========================================================================
__global__ void lstm_prep_weights(const float* __restrict__ W,
                                  const float* __restrict__ W_out,
                                  bf16_t* __restrict__ Wxf,
                                  bf16_t* __restrict__ Whf,
                                  bf16_t* __restrict__ Woutf,
                                  int* __restrict__ cnt) {
    const int e = blockIdx.x * blockDim.x + threadIdx.x;   // 0 .. 262143
    if (e == 0) *cnt = 0;
    if (e >= G4 * 8 * 32) return;
    const int kk = e & 31;
    const int kc = (e >> 5) & 7;
    const int n  = e >> 8;
    const int k  = kc * 32 + (kk & 7) + ((kk >> 3) & 1) * 16 + ((kk >> 4) & 1) * 8;
    Wxf[e] = (__bf16)W[(size_t)k * G4 + n];                 // rows 0..255: W_x
    Whf[e] = (__bf16)W[(size_t)(D_ + k) * G4 + n];          // rows 256..511: W_h
    if (e < 32 * 8 * 32) {
        const int n2 = e >> 8;
        Woutf[e] = (n2 < A_) ? (__bf16)W_out[(size_t)k * A_ + n2] : (__bf16)0.0f;
    }
}

// ===========================================================================
// Kernel 2: G[b*T+t, :] = x[b,t,:] @ Wx + bias   (parallel, 65536x256x1024)
// ===========================================================================
__global__ __launch_bounds__(256) void lstm_xgemm(const float* __restrict__ x,
                                                  const bf16_t* __restrict__ Wxf,
                                                  const float* __restrict__ bias,
                                                  float* __restrict__ G) {
    const int lane = threadIdx.x & 31;
    const int wid  = blockIdx.x * (blockDim.x >> 5) + (threadIdx.x >> 5);
    const int row0 = (wid >> 4) * 16;   // 16 rows of B*T
    const int col0 = (wid & 15) * 64;   // 64-col strip of 1024

    v8f acc[4];
#pragma unroll
    for (int nt = 0; nt < 4; ++nt) {
        const float bv = bias[col0 + nt * 16 + (lane & 15)];
#pragma unroll
        for (int r = 0; r < 8; ++r) acc[nt][r] = bv;
    }
#pragma unroll
    for (int kc = 0; kc < 8; ++kc) {
        const v16bf a = load_a_f32(x, D_, row0, kc * 32, lane);
#pragma unroll
        for (int nt = 0; nt < 4; ++nt) {
            const v16bf bf = load_b_frag(Wxf, col0 + nt * 16, kc, lane);
            acc[nt] = wmma_bf16(a, bf, acc[nt]);
        }
    }
    const int mrow = row0 + ((lane >> 4) & 1) * 8;
    const int nlo  = lane & 15;
#pragma unroll
    for (int nt = 0; nt < 4; ++nt)
#pragma unroll
        for (int r = 0; r < 8; ++r)
            G[(size_t)(mrow + r) * G4 + col0 + nt * 16 + nlo] = acc[nt][r];
}

// ===========================================================================
// Kernel 3: recurrence on 16 persistent WGs (256 thr / 8 waves each).
// WG w owns h-cols [16w,16w+16) and gate cols {16w+g*256}: cell update is
// WG-local. Per step: 8 WMMA/wave; h (bf16, double-buffered in global/L2)
// is the only cross-WG state -> ONE grid barrier per step.
// Gate slab (8KB/WG/step) is async-prefetched to LDS one step ahead.
// ===========================================================================
__global__ __launch_bounds__(256) void lstm_recur16(const float* __restrict__ G,
                                                    const bf16_t* __restrict__ Whf,
                                                    const int*   __restrict__ seq_lens,
                                                    const float* __restrict__ c_in,
                                                    const float* __restrict__ h_in,
                                                    bf16_t* __restrict__ Hg,   // [2][32][256]
                                                    int* __restrict__ cnt,
                                                    float* __restrict__ last_layer,
                                                    float* __restrict__ c_fin,
                                                    float* __restrict__ h_fin) {
    __shared__ float gateS[B_][64];        // 8 KB  : this step's gates (i|j|f|o slabs)
    __shared__ float gbuf[2][B_][64];      // 16 KB : async-prefetched x-gate slabs

    const int w    = blockIdx.x;           // 0..15 : column-slab owner
    const int tid  = threadIdx.x;
    const int lane = tid & 31;
    const int v    = tid >> 5;             // wave 0..7
    const int mt   = (v & 1) * 16;         // batch-row tile base
    const int g    = v >> 1;               // gate block 0..3 (i,j,f,o)
    const int nlo  = lane & 15;
    const int mhi8 = ((lane >> 4) & 1) * 8;
    const int nW   = g * 256 + w * 16;     // global gate-column base of this wave

    // elementwise ownership: 512 elems / 256 threads = 2 each
    const int eb  = tid >> 3;              // batch 0..31
    const int j0  = (tid & 7) * 2;         // 2 consecutive owned columns
    const int hc0 = w * 16 + j0;

    float c_reg[2], h_reg[2];
#pragma unroll
    for (int i = 0; i < 2; ++i) {
        c_reg[i] = c_in[(size_t)eb * H_ + hc0 + i];
        h_reg[i] = h_in[(size_t)eb * H_ + hc0 + i];
        Hg[(size_t)eb * H_ + hc0 + i] = (__bf16)h_reg[i];   // Hg[0] init (own cols)
    }
    const int sl = seq_lens[eb];

    // async prefetch chunk mapping: 512 x 16B chunks per step, 2 per thread
    int   ldsoff[2];
    size_t gbase[2];
#pragma unroll
    for (int i = 0; i < 2; ++i) {
        const int c = tid + i * 256;       // chunk id 0..511
        const int b = c >> 4;              // batch row
        const int q = c & 15;              // 16B chunk within 64-float row
        ldsoff[i] = b * 64 + q * 4;        // float offset in [32][64] slab
        gbase[i]  = ((size_t)b * T_) * G4 + (size_t)(q >> 2) * 256 + w * 16 + (q & 3) * 4;
    }
    // prologue: prefetch t=0 slab into gbuf[0]
#pragma unroll
    for (int i = 0; i < 2; ++i)
        async_g2l_b128(&gbuf[0][0][0] + ldsoff[i], G + gbase[i]);
    wait_async0();
    int phase = 1;
    grid_barrier(cnt, NWG * phase); ++phase;

    for (int t = 0; t < T_; ++t) {
        const int p = t & 1;

        // gates = (prefetched x-part) ...
        v8f acc;
#pragma unroll
        for (int r = 0; r < 8; ++r)
            acc[r] = gbuf[p][mt + mhi8 + r][g * 16 + nlo];

        // ... kick off next step's slab prefetch (overlaps the WMMAs)
        if (t + 1 < T_) {
#pragma unroll
            for (int i = 0; i < 2; ++i)
                async_g2l_b128(&gbuf[p ^ 1][0][0] + ldsoff[i],
                               G + gbase[i] + (size_t)(t + 1) * G4);
        }

        // ... + h @ W_h  (A from L2-hot Hg, B from fragment-packed Whf)
        const bf16_t* Hsrc = Hg + (size_t)p * (B_ * H_);
#pragma unroll
        for (int kc = 0; kc < 8; ++kc) {
            const v16bf a  = load_a_bf16(Hsrc, H_, mt, kc * 32, lane);
            const v16bf bf = load_b_frag(Whf, nW, kc, lane);
            acc = wmma_bf16(a, bf, acc);
        }

        // scatter gate tile into WG-local exchange
#pragma unroll
        for (int r = 0; r < 8; ++r)
            gateS[mt + mhi8 + r][g * 16 + nlo] = acc[r];
        __syncthreads();

        // elementwise LSTM cell (f32), fully WG-local
        const bool live = (t < sl);
        bf16_t* Hdst = Hg + (size_t)(p ^ 1) * (B_ * H_);
#pragma unroll
        for (int i = 0; i < 2; ++i) {
            const int j  = j0 + i;
            const int hc = w * 16 + j;
            const float gi = gateS[eb][j];
            const float gj = gateS[eb][16 + j];
            const float gf = gateS[eb][32 + j];
            const float go = gateS[eb][48 + j];
            const float nc = c_reg[i] * sigmoidf_(gf + 1.0f) +
                             sigmoidf_(gi) * tanhf(gj);
            const float nh = tanhf(nc) * sigmoidf_(go);
            if (live) { c_reg[i] = nc; h_reg[i] = nh; }
            last_layer[((size_t)eb * T_ + t) * H_ + hc] = live ? nh : 0.0f;
            Hdst[(size_t)eb * H_ + hc] = (__bf16)h_reg[i];
        }
        __threadfence();        // publish Hdst before the step barrier
        wait_async0();          // next slab resident before anyone reads it
        grid_barrier(cnt, NWG * phase); ++phase;
    }

#pragma unroll
    for (int i = 0; i < 2; ++i) {
        c_fin[(size_t)eb * H_ + hc0 + i] = c_reg[i];
        h_fin[(size_t)eb * H_ + hc0 + i] = h_reg[i];
    }
}

// ===========================================================================
// Kernel 4: logits = last_layer @ W_out + b_out   (N=18 padded to 2x16 tiles)
// ===========================================================================
__global__ __launch_bounds__(256) void lstm_proj(const float* __restrict__ last_layer,
                                                 const bf16_t* __restrict__ Woutf,
                                                 const float* __restrict__ b_out,
                                                 float* __restrict__ logits) {
    const int lane = threadIdx.x & 31;
    const int wid  = blockIdx.x * (blockDim.x >> 5) + (threadIdx.x >> 5);
    const int row0 = wid * 16;
    const int nlo  = lane & 15;

    v8f acc0, acc1;
    const float bv0 = b_out[nlo];
    const float bv1 = (16 + nlo < A_) ? b_out[16 + nlo] : 0.0f;
#pragma unroll
    for (int r = 0; r < 8; ++r) { acc0[r] = bv0; acc1[r] = bv1; }

#pragma unroll
    for (int kc = 0; kc < 8; ++kc) {
        const v16bf a  = load_a_f32(last_layer, H_, row0, kc * 32, lane);
        const v16bf b0 = load_b_frag(Woutf, 0,  kc, lane);
        const v16bf b1 = load_b_frag(Woutf, 16, kc, lane);
        acc0 = wmma_bf16(a, b0, acc0);
        acc1 = wmma_bf16(a, b1, acc1);
    }

    const int mrow = row0 + ((lane >> 4) & 1) * 8;
#pragma unroll
    for (int r = 0; r < 8; ++r)
        logits[(size_t)(mrow + r) * A_ + nlo] = acc0[r];
    if (16 + nlo < A_) {
#pragma unroll
        for (int r = 0; r < 8; ++r)
            logits[(size_t)(mrow + r) * A_ + 16 + nlo] = acc1[r];
    }
}

// ===========================================================================
extern "C" void kernel_launch(void* const* d_in, const int* in_sizes, int n_in,
                              void* d_out, int out_size, void* d_ws, size_t ws_size,
                              hipStream_t stream) {
    const float* x        = (const float*)d_in[0];
    const int*   seq_lens = (const int*)  d_in[1];
    const float* c_in     = (const float*)d_in[2];
    const float* h_in     = (const float*)d_in[3];
    const float* W        = (const float*)d_in[4];
    const float* b        = (const float*)d_in[5];
    const float* W_out    = (const float*)d_in[6];
    const float* b_out    = (const float*)d_in[7];

    // Workspace layout
    char* ws = (char*)d_ws;
    bf16_t* Wxf   = (bf16_t*)ws;                                   // 512 KB
    bf16_t* Whf   = (bf16_t*)(ws + (512ull << 10));                // 512 KB
    bf16_t* Woutf = (bf16_t*)(ws + (1024ull << 10));               //  16 KB
    bf16_t* Hg    = (bf16_t*)(ws + (1040ull << 10));               //  32 KB (h, x2 buf)
    float*  G     = (float*) (ws + (1072ull << 10));               // 256 MB
    int*    cnt   = (int*)   (ws + (1072ull << 10) + (size_t)B_ * T_ * G4 * 4);

    // Output layout (flat concat in reference return order)
    float* out        = (float*)d_out;
    float* logits     = out;                                       // B*T*A
    float* last_layer = logits + (size_t)B_ * T_ * A_;             // B*T*H
    float* c_fin      = last_layer + (size_t)B_ * T_ * H_;         // B*H
    float* h_fin      = c_fin + (size_t)B_ * H_;                   // B*H

    lstm_prep_weights<<<(G4 * 8 * 32) / 256, 256, 0, stream>>>(W, W_out, Wxf, Whf, Woutf, cnt);
    lstm_xgemm<<<8192, 256, 0, stream>>>(x, Wxf, b, G);
    lstm_recur16<<<NWG, 256, 0, stream>>>(G, Whf, seq_lens, c_in, h_in, Hg, cnt,
                                          last_layer, c_fin, h_fin);
    lstm_proj<<<512, 256, 0, stream>>>(last_layer, Woutf, b_out, logits);
}
